// MultiHeadSelfAttention_41747082117395
// MI455X (gfx1250) — matrix operand order repaired and checked by reading, not compiled, not run on hardware
//
#include <hip/hip_runtime.h>

typedef _Float16 half16 __attribute__((ext_vector_type(16)));
typedef _Float16 half8  __attribute__((ext_vector_type(8)));
typedef float    float8 __attribute__((ext_vector_type(8)));

#define NHEAD 16
#define DK    64
#define DMOD  1024
#define SEQ   2048
#define BATCH 2

#define WMMA_F16(A, B, C) \
    __builtin_amdgcn_wmma_f32_16x16x32_f16(false, (A), false, (B), (short)0, (C), false, false)

// ---------------------------------------------------------------------------
// Prep: W [h][d][e] f32  ->  Wt [h][e][d] f16  (B-fragment friendly).
// Folds the 1/sqrt(DK)=0.125 attention scale into Wq.
// ---------------------------------------------------------------------------
__global__ __launch_bounds__(256)
void prep_w(const float* __restrict__ Wq, const float* __restrict__ Wk,
            const float* __restrict__ Wv,
            _Float16* __restrict__ tq, _Float16* __restrict__ tk,
            _Float16* __restrict__ tv)
{
    int idx = blockIdx.x * 256 + threadIdx.x;     // 0 .. 16*64*64-1
    int d = idx & 63;
    int e = (idx >> 6) & 63;
    int h = idx >> 12;
    int src = (h * 64 + d) * 64 + e;
    tq[idx] = (_Float16)(0.125f * Wq[src]);
    tk[idx] = (_Float16)Wk[src];
    tv[idx] = (_Float16)Wv[src];
}

// Build a 16x32 f16 A-fragment from a row-major f32 row (one row per lane&15).
// elem j -> K = dbase+hi*16+j
__device__ inline half16 load_a_from_f32(const float* __restrict__ row, int dbase, int hi)
{
    float8 lo = *(const float8*)(row + dbase + hi * 16);
    float8 up = *(const float8*)(row + dbase + hi * 16 + 8);
    half16 a;
#pragma unroll
    for (int i = 0; i < 8; ++i) { a[i] = (_Float16)lo[i]; a[8 + i] = (_Float16)up[i]; }
    return a;
}

// ---------------------------------------------------------------------------
// WMMA projection: one wave per (b, h, 16-row tile).
//   D(16s x 16e) = X(16s x 64d) . W(64d x 64e) + bias   (bias via C init:
//   bias depends only on the column -> identical across the 8 C rows/lane)
// TRANS=false: store row-major [head][s][d];  TRANS=true: store [head][d][s].
// ---------------------------------------------------------------------------
template <bool TRANS>
__device__ inline void project_one(const float* __restrict__ xrow,
                                   const _Float16* __restrict__ wbase, // + h*64*64
                                   const float* __restrict__ bias,     // + h*64
                                   float bscale, int n, int hi, int s0, int head,
                                   _Float16* __restrict__ dst)
{
    half16 a0 = load_a_from_f32(xrow, 0, hi);
    half16 a1 = load_a_from_f32(xrow, 32, hi);
#pragma unroll
    for (int c = 0; c < 4; ++c) {
        const _Float16* wrow = wbase + (size_t)(c * 16 + n) * DK;
        half16 w0 = *(const half16*)(wrow + hi * 16);
        half16 w1 = *(const half16*)(wrow + 32 + hi * 16);
        float bv = bscale * bias[c * 16 + n];
        float8 acc;
#pragma unroll
        for (int r = 0; r < 8; ++r) acc[r] = bv;
        acc = WMMA_F16(a0, w0, acc);
        acc = WMMA_F16(a1, w1, acc);
        if (TRANS) {
            half8 pk;
#pragma unroll
            for (int r = 0; r < 8; ++r) pk[r] = (_Float16)acc[r];
            *(half8*)(dst + ((size_t)head * DK + c * 16 + n) * SEQ + s0 + hi * 8) = pk;
        } else {
#pragma unroll
            for (int r = 0; r < 8; ++r)
                dst[((size_t)head * SEQ + s0 + hi * 8 + r) * DK + c * 16 + n] = (_Float16)acc[r];
        }
    }
}

__global__ __launch_bounds__(32)
void proj_wmma(const float* __restrict__ Q, const float* __restrict__ K,
               const float* __restrict__ V,
               const _Float16* __restrict__ tWq, const float* __restrict__ bq,
               const _Float16* __restrict__ tWk, const float* __restrict__ bk,
               const _Float16* __restrict__ tWv, const float* __restrict__ bv,
               _Float16* __restrict__ qh, _Float16* __restrict__ kh,
               _Float16* __restrict__ vT)
{
    const int lane = threadIdx.x;
    const int n    = lane & 15;
    const int hi   = lane >> 4;
    const int s0   = blockIdx.x * 16;
    const int h    = blockIdx.y;
    const int b    = blockIdx.z;
    const int head = b * NHEAD + h;

    const size_t xoff = ((size_t)(b * SEQ + s0 + n)) * DMOD + (size_t)h * DK;
    project_one<false>(Q + xoff, tWq + (size_t)h * DK * DK, bq + h * DK, 0.125f, n, hi, s0, head, qh);
    project_one<false>(K + xoff, tWk + (size_t)h * DK * DK, bk + h * DK, 1.0f,   n, hi, s0, head, kh);
    project_one<true >(V + xoff, tWv + (size_t)h * DK * DK, bv + h * DK, 1.0f,   n, hi, s0, head, vT);
}

// ---------------------------------------------------------------------------
// Flash attention with TRANSPOSED scores: one wave per (b,h, 16-query tile).
//   S^T(32k x 16q) = K(32k x 64d) . Q^T(64d x 16q)    -> D[row=key][col=query]
//   => each lane holds 16 key-scores for ONE query (n=lane&15):
//      hi=0: keys {0..7} (tile0 regs) and {16..23} (tile1 regs)
//      hi=1: keys {8..15} and {24..31}
//   row max / row sum = 15 lane-local ops + ONE shfl_xor(16).
//   O^T(64d x 16q) += V^T(64d x 32k) . P^T(32k x 16q); A-frags come straight
//   from the vT layout, and P^T's B-fragment needs only 8 shfl_xor(16).
// ---------------------------------------------------------------------------
__global__ __launch_bounds__(32)
void attn_kernel(const _Float16* __restrict__ qh, const _Float16* __restrict__ kh,
                 const _Float16* __restrict__ vT, float* __restrict__ out)
{
    const int lane = threadIdx.x;
    const int n    = lane & 15;
    const int hi   = lane >> 4;
    const int qt   = blockIdx.x;      // 0..S/16-1
    const int bh   = blockIdx.y;      // 0..B*H-1
    const int b    = bh >> 4;
    const int h    = bh & 15;

    const _Float16* kbase = kh + (size_t)bh * SEQ * DK;
    const _Float16* vbase = vT + (size_t)bh * DK * SEQ;

    // Q^T B-fragments (resident for the whole loop):
    // b[j] = q[query n][d = dbase + hi*16 + j]
    const _Float16* qrow = qh + ((size_t)bh * SEQ + (size_t)qt * 16 + n) * DK;
    half16 bq0 = *(const half16*)(qrow + hi * 16);
    half16 bq1 = *(const half16*)(qrow + 32 + hi * 16);

    float8 o0 = {}, o1 = {}, o2 = {}, o3 = {};
    float mi = -1e30f, li = 0.0f;     // per-lane: softmax state of query n

    for (int kt = 0; kt < SEQ; kt += 32) {
        // ---- K A-fragments: a[j] = k[key][d], rows = keys ----
        const _Float16* krow0 = kbase + (size_t)(kt + n) * DK;
        const _Float16* krow1 = kbase + (size_t)(kt + 16 + n) * DK;
        half16 ka00, ka01, ka10, ka11;
        {
            half8 c0 = *(const half8*)(krow0 + hi * 16);
            half8 c1 = *(const half8*)(krow0 + 8 + hi * 16);
            half8 c2 = *(const half8*)(krow0 + 32 + hi * 16);
            half8 c3 = *(const half8*)(krow0 + 40 + hi * 16);
            half8 c4 = *(const half8*)(krow1 + hi * 16);
            half8 c5 = *(const half8*)(krow1 + 8 + hi * 16);
            half8 c6 = *(const half8*)(krow1 + 32 + hi * 16);
            half8 c7 = *(const half8*)(krow1 + 40 + hi * 16);
#pragma unroll
            for (int i = 0; i < 8; ++i) {
                ka00[i] = c0[i]; ka00[8 + i] = c1[i];
                ka01[i] = c2[i]; ka01[8 + i] = c3[i];
                ka10[i] = c4[i]; ka10[8 + i] = c5[i];
                ka11[i] = c6[i]; ka11[8 + i] = c7[i];
            }
        }

        float8 s0 = {}, s1 = {};
        s0 = WMMA_F16(ka00, bq0, s0);
        s0 = WMMA_F16(ka01, bq1, s0);
        s1 = WMMA_F16(ka10, bq0, s1);
        s1 = WMMA_F16(ka11, bq1, s1);

        // ---- online softmax: lane-local over 16 values + one cross-half shuffle
        float vmax = fmaxf(s0[0], s1[0]);
#pragma unroll
        for (int r = 1; r < 8; ++r) vmax = fmaxf(vmax, fmaxf(s0[r], s1[r]));
        vmax = fmaxf(vmax, __shfl_xor(vmax, 16, 32));

        float mnew = fmaxf(mi, vmax);
        float corr = __expf(mi - mnew);
        float rs = 0.0f;
#pragma unroll
        for (int r = 0; r < 8; ++r) {
            s0[r] = __expf(s0[r] - mnew);
            s1[r] = __expf(s1[r] - mnew);
            rs += s0[r] + s1[r];
        }
        rs += __shfl_xor(rs, 16, 32);
        li = li * corr + rs;
        mi = mnew;
#pragma unroll
        for (int r = 0; r < 8; ++r) {
            o0[r] *= corr; o1[r] *= corr; o2[r] *= corr; o3[r] *= corr;
        }

        // ---- P^T B-fragment: b[j] = P^T[key=hi*16+j][query n]
        // half-swap with paired lane (8 shuffles, no LDS)
        half16 pB;
#pragma unroll
        for (int r = 0; r < 8; ++r) {
            float send = hi ? s0[r] : s1[r];
            float recv = __shfl_xor(send, 16, 32);
            pB[r]     = (_Float16)(hi ? recv : s0[r]);   // j=r   -> key hi*16+r
            pB[8 + r] = (_Float16)(hi ? s1[r] : recv);   // j=8+r -> key hi*16+8+r
        }

        // ---- O^T += V^T . P^T : A-frags straight from vT layout ----
#pragma unroll
        for (int c = 0; c < 4; ++c) {
            const _Float16* vrow = vbase + (size_t)(c * 16 + n) * SEQ + kt;
            half16 va;
            {
                half8 c0 = *(const half8*)(vrow + hi * 16);
                half8 c1 = *(const half8*)(vrow + 8 + hi * 16);
#pragma unroll
                for (int i = 0; i < 8; ++i) { va[i] = c0[i]; va[8 + i] = c1[i]; }
            }
            if (c == 0) o0 = WMMA_F16(va, pB, o0);
            if (c == 1) o1 = WMMA_F16(va, pB, o1);
            if (c == 2) o2 = WMMA_F16(va, pB, o2);
            if (c == 3) o3 = WMMA_F16(va, pB, o3);
        }
    }

    // ---- normalize + transpose via LDS for coalesced stores ----
    __shared__ float ldsp[16 * 68];           // [query][64 d + pad]
    float inv = 1.0f / li;
#pragma unroll
    for (int r = 0; r < 8; ++r) {
        ldsp[n * 68 + 0  + hi * 8 + r] = o0[r] * inv;
        ldsp[n * 68 + 16 + hi * 8 + r] = o1[r] * inv;
        ldsp[n * 68 + 32 + hi * 8 + r] = o2[r] * inv;
        ldsp[n * 68 + 48 + hi * 8 + r] = o3[r] * inv;
    }
    // single wave: DS ops are in-order, no barrier needed
#pragma unroll 4
    for (int q = 0; q < 16; ++q) {
        float* orow = out + ((size_t)(b * SEQ + qt * 16 + q)) * DMOD + (size_t)h * DK;
        orow[lane]      = ldsp[q * 68 + lane];
        orow[32 + lane] = ldsp[q * 68 + 32 + lane];
    }
}

// ---------------------------------------------------------------------------
extern "C" void kernel_launch(void* const* d_in, const int* in_sizes, int n_in,
                              void* d_out, int out_size, void* d_ws, size_t ws_size,
                              hipStream_t stream) {
    (void)in_sizes; (void)n_in; (void)out_size; (void)ws_size;
    const float* Q  = (const float*)d_in[0];
    const float* K  = (const float*)d_in[1];
    const float* V  = (const float*)d_in[2];
    const float* Wq = (const float*)d_in[3];
    const float* bq = (const float*)d_in[4];
    const float* Wk = (const float*)d_in[5];
    const float* bk = (const float*)d_in[6];
    const float* Wv = (const float*)d_in[7];
    const float* bv = (const float*)d_in[8];
    float* out = (float*)d_out;

    const size_t elems = (size_t)BATCH * NHEAD * SEQ * DK;   // 4,194,304
    const size_t welem = (size_t)NHEAD * DK * DK;            // 65,536
    _Float16* qh  = (_Float16*)d_ws;        //  8 MiB
    _Float16* kh  = qh + elems;             //  8 MiB
    _Float16* vT  = kh + elems;             //  8 MiB
    _Float16* tWq = vT + elems;             //  128 KiB
    _Float16* tWk = tWq + welem;            //  128 KiB
    _Float16* tWv = tWk + welem;            //  128 KiB   (~24.4 MiB total)

    prep_w<<<(NHEAD * DK * DK) / 256, 256, 0, stream>>>(Wq, Wk, Wv, tWq, tWk, tWv);

    dim3 pgrid(SEQ / 16, NHEAD, BATCH);
    proj_wmma<<<pgrid, 32, 0, stream>>>(Q, K, V, tWq, bq, tWk, bk, tWv, bv, qh, kh, vT);

    dim3 agrid(SEQ / 16, BATCH * NHEAD, 1);
    attn_kernel<<<agrid, 32, 0, stream>>>(qh, kh, vT, out);
}
